// LWPCEAM_2293512536676
// MI455X (gfx1250) — matrix operand order repaired and checked by reading, not compiled, not run on hardware
//
#include <hip/hip_runtime.h>
#include <math.h>

// ---------------------------------------------------------------------------
// MI455X (gfx1250) implementation.
//  * b_q is mathematically exactly 1/N (softmax rows sum to 1) -> Q/K skipped.
//  * attended = bq * (patch @ Wout)         : bf16 WMMA, f32 accum
//  * tau_n    = min_m sqrt(q2 + m2 - 2 a.m) : bf16 WMMA cross term, exact f32
//               norms, ordered-uint atomic min reduction
//  * A tiles staged by TDM tensor_load_to_lds (TENSORcnt), B tiles by
//    global_load_async_to_lds_b128 (ASYNCcnt).
//  * heat -> bilinear 56->224 -> minmax normalize
// ---------------------------------------------------------------------------

#define NPATCH 3136
#define DDIM   512
#define MBANK  50000
#define GRID_HW 56
#define IMG    224
#define NT 64          // rows per workgroup tile
#define MT 32          // cols per workgroup tile
#define KC 64          // K chunk staged in LDS
#define SA 72          // LDS row stride (ushorts): 128B data + 16B pad
#define SB 72
#define GAIN_C 5.0f
#define THRESH_C 0.5f

typedef __attribute__((ext_vector_type(16))) __bf16 v16bf;
typedef __attribute__((ext_vector_type(8)))  float  v8f;
typedef __attribute__((ext_vector_type(4)))  unsigned v4u;
typedef __attribute__((ext_vector_type(8)))  int      v8i;
typedef __attribute__((ext_vector_type(4)))  int      v4i;

union FragBF { v16bf v; unsigned d[8]; };
union AccF   { v8f  v; float f[8]; };

__device__ __forceinline__ unsigned short f2bf(float f) {
  unsigned u = __float_as_uint(f);
  unsigned r = 0x7FFFu + ((u >> 16) & 1u);          // round-to-nearest-even
  return (unsigned short)((u + r) >> 16);
}
__device__ __forceinline__ float bf2f(unsigned short h) {
  return __uint_as_float(((unsigned)h) << 16);
}
// monotone float<->uint encode (total order incl. negatives)
__device__ __forceinline__ unsigned fenc(float f) {
  unsigned u = __float_as_uint(f);
  return (u & 0x80000000u) ? ~u : (u | 0x80000000u);
}
__device__ __forceinline__ float fdec(unsigned u) {
  return __uint_as_float((u & 0x80000000u) ? (u ^ 0x80000000u) : ~u);
}

// LDS aperture: flat address low 32 bits == LDS byte address (ISA 10.2)
__device__ __forceinline__ unsigned lds_addr32(const void* p) {
  return (unsigned)(uintptr_t)p;
}

// Per-lane async global->LDS B128 copy (ASYNCcnt).
__device__ __forceinline__ void async_g2l_b128(unsigned ldsoff, const void* g) {
  asm volatile("global_load_async_to_lds_b128 %0, %1, off"
               :: "v"(ldsoff), "v"((unsigned long long)(uintptr_t)g)
               : "memory");
}
__device__ __forceinline__ void wait_async0() {
  asm volatile("s_wait_asynccnt 0" ::: "memory");
}

// TDM: load a 64x64 bf16 tile (row stride DDIM elements) into LDS with
// 16B padding per 128B row (-> SA=72 ushort rows). One instruction per wave.
__device__ __forceinline__ void tdm_load_tile_64x64(const unsigned short* g,
                                                    unsigned ldsoff) {
  unsigned long long ga = (unsigned long long)(uintptr_t)g;
  v4u g0;
  g0.x = 1u;                                          // count=1, user mode
  g0.y = ldsoff;                                      // lds_addr
  g0.z = (unsigned)(ga & 0xFFFFFFFFu);                // global_addr[31:0]
  g0.w = (unsigned)((ga >> 32) & 0x01FFFFFFu)         // global_addr[56:32]
         | (2u << 30);                                // type=2 (image)
  v8i g1;
  g1[0] = (int)((1u << 16)                            // data_size = 2 bytes
              | (1u << 20)                            // pad_enable
              | (4u << 22)                            // pad_interval: 32 dwords
              | (3u << 25));                          // pad_amount : 4 dwords
  g1[1] = (int)(64u << 16);                           // tensor_dim0 = 64
  g1[2] = (int)(64u << 16);                           // tensor_dim1 = 64
  g1[3] = (int)(64u << 16);                           // tile_dim0   = 64
  g1[4] = (int)(64u);                                 // tile_dim1   = 64
  g1[5] = (int)(unsigned)DDIM;                        // tensor_dim0_stride
  g1[6] = 0;
  g1[7] = 0;
  v4i g2 = {0, 0, 0, 0};
  v4i g3 = {0, 0, 0, 0};
#if __clang_major__ >= 23
  v8i g4 = {0, 0, 0, 0, 0, 0, 0, 0};
  __builtin_amdgcn_tensor_load_to_lds(g0, g1, g2, g3, g4, 0);
#else
  __builtin_amdgcn_tensor_load_to_lds(g0, g1, g2, g3, 0);
#endif
}

// ---------------------------------------------------------------- converters
__global__ __launch_bounds__(256) void conv_mem_kernel(
    const float* __restrict__ M, unsigned short* __restrict__ Mbf,
    float* __restrict__ m2) {
  int row  = blockIdx.x * 8 + (threadIdx.x >> 5);
  int lane = threadIdx.x & 31;
  if (row >= MBANK) return;
  const float4* src = (const float4*)(M + (size_t)row * DDIM + lane * 16);
  unsigned short h[16];
  float s = 0.f;
#pragma unroll
  for (int i = 0; i < 4; ++i) {
    float4 v = src[i];
    s += v.x * v.x + v.y * v.y + v.z * v.z + v.w * v.w;
    h[4 * i + 0] = f2bf(v.x); h[4 * i + 1] = f2bf(v.y);
    h[4 * i + 2] = f2bf(v.z); h[4 * i + 3] = f2bf(v.w);
  }
  unsigned p[8];
#pragma unroll
  for (int i = 0; i < 8; ++i)
    p[i] = (unsigned)h[2 * i] | ((unsigned)h[2 * i + 1] << 16);
  uint4* dst = (uint4*)(Mbf + (size_t)row * DDIM + lane * 16);
  dst[0] = make_uint4(p[0], p[1], p[2], p[3]);
  dst[1] = make_uint4(p[4], p[5], p[6], p[7]);
#pragma unroll
  for (int sh = 1; sh < 32; sh <<= 1) s += __shfl_xor(s, sh, 32);
  if (lane == 0) m2[row] = s;
}

__global__ __launch_bounds__(256) void conv_patch_kernel(
    const float* __restrict__ P, unsigned short* __restrict__ Pbf) {
  int idx = blockIdx.x * 256 + threadIdx.x;           // NPATCH*DDIM exact
  Pbf[idx] = f2bf(P[idx]);
}

__global__ __launch_bounds__(256) void conv_woutT_kernel(
    const float* __restrict__ W, unsigned short* __restrict__ WT) {
  int idx = blockIdx.x * 256 + threadIdx.x;           // DDIM*DDIM exact
  int k = idx >> 9, j = idx & 511;                    // coalesced read
  WT[(size_t)j * DDIM + k] = f2bf(W[idx]);            // store W^T as bf16
}

__global__ void init_min_kernel(unsigned* __restrict__ minT) {
  int n = blockIdx.x * 256 + threadIdx.x;
  if (n < NPATCH) minT[n] = 0xFF800000u;              // fenc(+inf)
}

// ------------------------------------------------ GEMM1: attended = bq*(P@W)
__global__ __launch_bounds__(256) void attn_gemm_kernel(
    const unsigned short* __restrict__ Abf,   // patch_bf  [N][K]
    const unsigned short* __restrict__ Bbf,   // woutT_bf  [J][K]
    unsigned short* __restrict__ Cbf,         // attended_bf [N][J]
    float bq) {
  __shared__ unsigned short Asl[NT * SA];
  __shared__ unsigned short Bsl[MT * SB];
  const int tid = threadIdx.x;
  const int wave = tid >> 5, lane = tid & 31;
  const int half = lane >> 4, l16 = lane & 15;
  const int nsub = wave >> 1, msub = wave & 1;
  const int n0 = blockIdx.y * NT;
  const int j0 = blockIdx.x * MT;
  const unsigned aOff = lds_addr32(Asl);
  const unsigned bOff = lds_addr32(Bsl);

  AccF acc;
#pragma unroll
  for (int i = 0; i < 8; ++i) acc.f[i] = 0.f;

  for (int kc = 0; kc < DDIM; kc += KC) {
    __syncthreads();
    if (wave == 0)                              // TDM: whole 64x64 A tile
      tdm_load_tile_64x64(Abf + (size_t)n0 * DDIM + kc, aOff);
    { // B tile 32x64: one async B128 per thread
      int row = tid >> 3, cg = (tid & 7) << 3;
      async_g2l_b128(bOff + (unsigned)(row * SB + cg) * 2u,
                     Bbf + (size_t)(j0 + row) * DDIM + kc + cg);
    }
    wait_async0();
    if (wave == 0) __builtin_amdgcn_s_wait_tensorcnt(0);
    __syncthreads();
#pragma unroll
    for (int kk = 0; kk < KC; kk += 32) {
      FragBF a, b;
      const unsigned* Ar = (const unsigned*)(Asl + (nsub * 16 + l16) * SA);
      const unsigned* Br = (const unsigned*)(Bsl + (msub * 16 + l16) * SB);
#pragma unroll
      for (int i = 0; i < 8; ++i) {
        int ka = (half << 4) + (i << 1);
        a.d[i] = Ar[(kk + ka) >> 1];
        int kb = (half << 4) + (i << 1);
        b.d[i] = Br[(kk + kb) >> 1];
      }
      acc.v = __builtin_amdgcn_wmma_f32_16x16x32_bf16(
          false, a.v, false, b.v, (short)0, acc.v, false, false);
    }
  }
  // epilogue: scale by bq, store bf16
#pragma unroll
  for (int r = 0; r < 8; ++r) {
    int row = n0 + nsub * 16 + r + 8 * half;
    int col = j0 + msub * 16 + l16;
    Cbf[(size_t)row * DDIM + col] = f2bf(acc.f[r] * bq);
  }
}

// ----------------------------------------------- q2 = rownorm2(attended_bf)
__global__ __launch_bounds__(256) void q2_kernel(
    const unsigned short* __restrict__ Abf, float* __restrict__ q2) {
  int row  = blockIdx.x * 8 + (threadIdx.x >> 5);
  int lane = threadIdx.x & 31;
  const uint4* src = (const uint4*)(Abf + (size_t)row * DDIM + lane * 16);
  float s = 0.f;
#pragma unroll
  for (int i = 0; i < 2; ++i) {
    uint4 v = src[i];
    unsigned w[4] = {v.x, v.y, v.z, v.w};
#pragma unroll
    for (int j = 0; j < 4; ++j) {
      float lo = bf2f((unsigned short)(w[j] & 0xFFFFu));
      float hi = bf2f((unsigned short)(w[j] >> 16));
      s += lo * lo + hi * hi;
    }
  }
#pragma unroll
  for (int sh = 1; sh < 32; sh <<= 1) s += __shfl_xor(s, sh, 32);
  if (lane == 0) q2[row] = s;
}

// -------------------------- GEMM2: min_m ( m2[m] - 2*dot(attended_n, mem_m) )
__global__ __launch_bounds__(256) void score_gemm_kernel(
    const unsigned short* __restrict__ Abf,   // attended_bf [N][K]
    const unsigned short* __restrict__ Mbf,   // mem_bf      [M][K]
    const float* __restrict__ m2,
    unsigned* __restrict__ minT) {
  __shared__ unsigned short Asl[NT * SA];
  __shared__ unsigned short Bsl[MT * SB];
  __shared__ float m2s[MT];
  __shared__ unsigned wgMin[NT];

  const int tid = threadIdx.x;
  const int wave = tid >> 5, lane = tid & 31;
  const int half = lane >> 4, l16 = lane & 15;
  const int nsub = wave >> 1, msub = wave & 1;
  const int n0  = blockIdx.y * NT;
  const int mg0 = blockIdx.x * 16;            // group of 16 m-tiles
  const unsigned aOff = lds_addr32(Asl);
  const unsigned bOff = lds_addr32(Bsl);

  if (tid < NT) wgMin[tid] = 0xFF800000u;     // fenc(+inf)
  __syncthreads();

  for (int it = 0; it < 16; ++it) {
    int m0 = (mg0 + it) * MT;
    if (m0 >= MBANK) break;                   // uniform exit
    if (tid < MT) {
      int m = m0 + tid;
      m2s[tid] = (m < MBANK) ? m2[m] : __uint_as_float(0x7F800000u);
    }
    AccF acc;
#pragma unroll
    for (int i = 0; i < 8; ++i) acc.f[i] = 0.f;

    for (int kc = 0; kc < DDIM; kc += KC) {
      __syncthreads();
      if (wave == 0)                          // TDM: 64x64 attended tile
        tdm_load_tile_64x64(Abf + (size_t)n0 * DDIM + kc, aOff);
      { // B tile 32x64 from memory bank: async B128 per thread (+prefetch)
        int row = tid >> 3, cg = (tid & 7) << 3;
        int m = m0 + row;
        if (m < MBANK) {
          const unsigned short* gp = Mbf + (size_t)m * DDIM + kc + cg;
          async_g2l_b128(bOff + (unsigned)(row * SB + cg) * 2u, gp);
          if (kc + KC < DDIM) __builtin_prefetch(gp + KC, 0, 1);
        } else {
          *(uint4*)(Bsl + row * SB + cg) = make_uint4(0u, 0u, 0u, 0u);
        }
      }
      wait_async0();
      if (wave == 0) __builtin_amdgcn_s_wait_tensorcnt(0);
      __syncthreads();
#pragma unroll
      for (int kk = 0; kk < KC; kk += 32) {
        FragBF a, b;
        const unsigned* Ar = (const unsigned*)(Asl + (nsub * 16 + l16) * SA);
        const unsigned* Br = (const unsigned*)(Bsl + (msub * 16 + l16) * SB);
#pragma unroll
        for (int i = 0; i < 8; ++i) {
          int ka = (half << 4) + (i << 1);
          a.d[i] = Ar[(kk + ka) >> 1];
          int kb = (half << 4) + (i << 1);
          b.d[i] = Br[(kk + kb) >> 1];
        }
        acc.v = __builtin_amdgcn_wmma_f32_16x16x32_bf16(
            false, a.v, false, b.v, (short)0, acc.v, false, false);
      }
    }
    // epilogue: t = m2 - 2*dot ; butterfly min over the 16 columns
    float m2v = m2s[msub * 16 + l16];
#pragma unroll
    for (int r = 0; r < 8; ++r) {
      float t = fmaf(-2.0f, acc.f[r], m2v);
#pragma unroll
      for (int sh = 1; sh < 16; sh <<= 1)
        t = fminf(t, __shfl_xor(t, sh, 32));
      if (l16 == 0)
        atomicMin(&wgMin[nsub * 16 + r + 8 * half], fenc(t));
    }
    __syncthreads();   // protect m2s before next iteration rewrites it
  }
  if (tid < NT) atomicMin(&minT[n0 + tid], wgMin[tid]);
}

// --------------------------------------------------------------- heat / map
__global__ void heat_kernel(const unsigned* __restrict__ minT,
                            const float* __restrict__ q2,
                            float* __restrict__ heat, float bqn) {
  int n = blockIdx.x * 256 + threadIdx.x;
  if (n >= NPATCH) return;
  float d2 = fmaxf(q2[n] + fdec(minT[n]), 0.f);
  float d = sqrtf(d2);
  float act = 1.f / (1.f + __expf(-GAIN_C * (d - THRESH_C)));
  heat[n] = act * (bqn + 1e-8f);
}

__global__ __launch_bounds__(256) void resize_kernel(
    const float* __restrict__ heat, float* __restrict__ up,
    float* __restrict__ pmin, float* __restrict__ pmax) {
  int idx = blockIdx.x * 256 + threadIdx.x;       // 224*224 = 196*256
  int oy = idx / IMG, ox = idx % IMG;
  float yy = (oy + 0.5f) * 0.25f - 0.5f;
  float xx = (ox + 0.5f) * 0.25f - 0.5f;
  int y0 = (int)floorf(yy), x0 = (int)floorf(xx);
  float fy = yy - y0, fx = xx - x0;
  int y0c = min(max(y0, 0), GRID_HW - 1), y1c = min(max(y0 + 1, 0), GRID_HW - 1);
  int x0c = min(max(x0, 0), GRID_HW - 1), x1c = min(max(x0 + 1, 0), GRID_HW - 1);
  float v00 = heat[y0c * GRID_HW + x0c], v01 = heat[y0c * GRID_HW + x1c];
  float v10 = heat[y1c * GRID_HW + x0c], v11 = heat[y1c * GRID_HW + x1c];
  float v = v00 * (1 - fy) * (1 - fx) + v01 * (1 - fy) * fx +
            v10 * fy * (1 - fx) + v11 * fy * fx;
  up[idx] = v;

  __shared__ float smn[256], smx[256];
  smn[threadIdx.x] = v; smx[threadIdx.x] = v;
  __syncthreads();
  for (int s = 128; s > 0; s >>= 1) {
    if (threadIdx.x < s) {
      smn[threadIdx.x] = fminf(smn[threadIdx.x], smn[threadIdx.x + s]);
      smx[threadIdx.x] = fmaxf(smx[threadIdx.x], smx[threadIdx.x + s]);
    }
    __syncthreads();
  }
  if (threadIdx.x == 0) { pmin[blockIdx.x] = smn[0]; pmax[blockIdx.x] = smx[0]; }
}

__global__ __launch_bounds__(256) void norm_kernel(
    const float* __restrict__ up, const float* __restrict__ pmin,
    const float* __restrict__ pmax, float* __restrict__ out) {
  __shared__ float smn[256], smx[256];
  int t = threadIdx.x;
  float mn = __uint_as_float(0x7F800000u), mx = __uint_as_float(0xFF800000u);
  if (t < 196) { mn = pmin[t]; mx = pmax[t]; }
  smn[t] = mn; smx[t] = mx;
  __syncthreads();
  for (int s = 128; s > 0; s >>= 1) {
    if (t < s) {
      smn[t] = fminf(smn[t], smn[t + s]);
      smx[t] = fmaxf(smx[t], smx[t + s]);
    }
    __syncthreads();
  }
  mn = smn[0]; mx = smx[0];
  int idx = blockIdx.x * 256 + t;
  float v = up[idx];
  out[idx] = (mx - mn > 1e-8f) ? (v - mn) / (mx - mn + 1e-12f) : v;
}

// ---------------------------------------------------------------- launcher
extern "C" void kernel_launch(void* const* d_in, const int* in_sizes, int n_in,
                              void* d_out, int out_size, void* d_ws,
                              size_t ws_size, hipStream_t stream) {
  const float* patch = (const float*)d_in[0];   // (3136,512)
  const float* mem   = (const float*)d_in[1];   // (50000,512)
  // d_in[2]=Wq, d_in[3]=Wk intentionally unused: softmax rows sum to 1
  // exactly, so b_q == 1/N independent of attention weights.
  const float* wout  = (const float*)d_in[4];   // (512,512)
  float* out = (float*)d_out;                   // (224,224)

  char* ws = (char*)d_ws;
  size_t off = 0;
  auto alloc = [&](size_t bytes) {
    size_t o = off;
    off += (bytes + 255) & ~(size_t)255;
    return o;
  };
  unsigned short* mem_bf   = (unsigned short*)(ws + alloc((size_t)MBANK * DDIM * 2));
  unsigned short* patch_bf = (unsigned short*)(ws + alloc((size_t)NPATCH * DDIM * 2));
  unsigned short* woutT_bf = (unsigned short*)(ws + alloc((size_t)DDIM * DDIM * 2));
  unsigned short* att_bf   = (unsigned short*)(ws + alloc((size_t)NPATCH * DDIM * 2));
  float*    m2   = (float*)(ws + alloc((size_t)MBANK * 4));
  float*    q2   = (float*)(ws + alloc((size_t)NPATCH * 4));
  unsigned* minT = (unsigned*)(ws + alloc((size_t)NPATCH * 4));
  float*    heat = (float*)(ws + alloc((size_t)NPATCH * 4));
  float*    up   = (float*)(ws + alloc((size_t)IMG * IMG * 4));
  float*    pmin = (float*)(ws + alloc(196 * 4));
  float*    pmax = (float*)(ws + alloc(196 * 4));
  (void)ws_size; (void)in_sizes; (void)n_in; (void)out_size;

  const float bqn = (1.0f / (float)NPATCH) / (1.0f + 1e-8f);

  conv_mem_kernel<<<MBANK / 8, 256, 0, stream>>>(mem, mem_bf, m2);
  conv_patch_kernel<<<(NPATCH * DDIM) / 256, 256, 0, stream>>>(patch, patch_bf);
  conv_woutT_kernel<<<(DDIM * DDIM) / 256, 256, 0, stream>>>(wout, woutT_bf);
  init_min_kernel<<<(NPATCH + 255) / 256, 256, 0, stream>>>(minT);

  attn_gemm_kernel<<<dim3(DDIM / MT, NPATCH / NT), 256, 0, stream>>>(
      patch_bf, woutT_bf, att_bf, bqn);
  q2_kernel<<<NPATCH / 8, 256, 0, stream>>>(att_bf, q2);

  // m-tiles = ceil(50000/32) = 1563, grouped 16 per WG -> 98 groups
  score_gemm_kernel<<<dim3(98, NPATCH / NT), 256, 0, stream>>>(
      att_bf, mem_bf, m2, minT);

  heat_kernel<<<(NPATCH + 255) / 256, 256, 0, stream>>>(minT, q2, heat, bqn);
  resize_kernel<<<(IMG * IMG) / 256, 256, 0, stream>>>(heat, up, pmin, pmax);
  norm_kernel<<<(IMG * IMG) / 256, 256, 0, stream>>>(up, pmin, pmax, out);
}
